// PoetryModel_38757784879521
// MI455X (gfx1250) — matrix operand order repaired and checked by reading, not compiled, not run on hardware
//
#include <hip/hip_runtime.h>

// ---------------------------------------------------------------------------
// LSTM poetry model, MI455X (gfx1250, wave32, WMMA).
// bf16 operands / fp32 accumulation via v_wmma_f32_16x16x32_bf16.
// Output GEMM: LDS-staged B panels via async global->LDS copy (ASYNCcnt).
// Recurrent kernel: h state lives in LDS (64 KB) inside one persistent WGP.
// ---------------------------------------------------------------------------

typedef __attribute__((ext_vector_type(16))) __bf16 v16bf;
typedef __attribute__((ext_vector_type(8)))  float  v8f;
typedef int v4i __attribute__((vector_size(16)));   // matches builtin param type

#define TT 128   // timesteps
#define BB 64    // batch
#define EE 256   // embed
#define HH 512   // hidden
#define G4 2048  // 4*hidden (i|f|c|o fused)
#define VV 10000 // vocab

#if defined(__has_builtin)
#if __has_builtin(__builtin_amdgcn_global_load_async_to_lds_b128)
#define HAVE_ASYNC_COPY 1
#endif
#if __has_builtin(__builtin_amdgcn_s_wait_asynccnt)
#define HAVE_ASYNC_WAIT 1
#endif
#endif

union Frag16 {
    v16bf v;
    uint4 q[2];
};

// A-fragment (16x32 bf16, row-major source, lda in elements).
// lanes 0-15: row M=m0+lane, K = k0..k0+15 ; lanes 16-31: K = k0+16..k0+31.
__device__ __forceinline__ v16bf load_a_frag(const __bf16* A, int lda, int m0, int k0) {
    const int lane = threadIdx.x & 31;
    const int row  = m0 + (lane & 15);
    const int kb   = k0 + ((lane >> 4) << 4);
    const __bf16* p = A + (size_t)row * lda + kb;
    Frag16 f;
    f.q[0] = *reinterpret_cast<const uint4*>(p);
    f.q[1] = *reinterpret_cast<const uint4*>(p + 8);
    return f.v;
}

// B-fragment (32x16 bf16) from a TRANSPOSED weight matrix BT[N][K] (ldb = K).
// lanes 0-15: col N=n0+lane, K = k0..k0+15 ; lanes 16-31: K = k0+16..k0+31.
__device__ __forceinline__ v16bf load_b_frag(const __bf16* BT, int ldb, int k0, int n0) {
    const int lane = threadIdx.x & 31;
    const int col  = n0 + (lane & 15);
    const int kb   = k0 + ((lane >> 4) << 4);
    const __bf16* p = BT + (size_t)col * ldb + kb;
    Frag16 f;
    f.q[0] = *reinterpret_cast<const uint4*>(p);
    f.q[1] = *reinterpret_cast<const uint4*>(p + 8);
    return f.v;
}

__device__ __forceinline__ v8f wmma_bf16(v16bf a, v16bf b, v8f c) {
    return __builtin_amdgcn_wmma_f32_16x16x32_bf16(
        false, a, false, b, (short)0, c, false, false);
}

// C/D layout: lanes 0-15 -> col N=n0+lane, rows m0+0..7 in regs 0..7;
//             lanes 16-31 -> same cols, rows m0+8..15.
__device__ __forceinline__ void store_c_tile(float* C, int ldc, int m0, int n0, v8f acc) {
    const int lane = threadIdx.x & 31;
    const int col  = n0 + (lane & 15);
    const int rb   = m0 + ((lane >> 4) << 3);
#pragma unroll
    for (int r = 0; r < 8; ++r)
        C[(size_t)(rb + r) * ldc + col] = acc[r];
}

// ---- async global -> LDS 16-byte copy (gfx1250 GLOBAL_LOAD_ASYNC_TO_LDS_B128)
__device__ __forceinline__ void copy16_to_lds(__bf16* lds_dst, const __bf16* gsrc) {
#if HAVE_ASYNC_COPY
    __builtin_amdgcn_global_load_async_to_lds_b128(
        (__attribute__((address_space(1))) v4i*)gsrc,
        (__attribute__((address_space(3))) v4i*)lds_dst,
        0, 0);
#else
    *reinterpret_cast<uint4*>(lds_dst) = *reinterpret_cast<const uint4*>(gsrc);
#endif
}

__device__ __forceinline__ void async_wait_le(int tag /*0 or 2*/) {
#if HAVE_ASYNC_COPY
#if HAVE_ASYNC_WAIT
    if (tag == 0) __builtin_amdgcn_s_wait_asynccnt(0);
    else          __builtin_amdgcn_s_wait_asynccnt(2);
#else
    if (tag == 0) asm volatile("s_wait_asynccnt 0x0" ::: "memory");
    else          asm volatile("s_wait_asynccnt 0x2" ::: "memory");
#endif
#endif
}

// ---------------------------------------------------------------------------
// Prep kernels: embedding gather -> bf16, fp32 weights -> transposed bf16.
// ---------------------------------------------------------------------------
__global__ void gather_embed_bf16(const int* __restrict__ ids,
                                  const float* __restrict__ emb,
                                  __bf16* __restrict__ out) {
    int idx = blockIdx.x * blockDim.x + threadIdx.x;
    if (idx >= TT * BB * EE) return;
    int r = idx / EE, e = idx - r * EE;
    out[idx] = (__bf16)emb[(size_t)ids[r] * EE + e];
}

// src[K][N] fp32 -> dst[N][K] bf16
__global__ void transpose_bf16(const float* __restrict__ src,
                               __bf16* __restrict__ dst, int K, int N) {
    int idx = blockIdx.x * blockDim.x + threadIdx.x;
    if (idx >= K * N) return;
    int n = idx / K, k = idx - n * K;
    dst[idx] = (__bf16)src[(size_t)k * N + n];
}

// ---------------------------------------------------------------------------
// Persistent recurrent kernel: one workgroup (32 wave32s) on one WGP.
// h state (64x512 bf16 = 64 KB) lives in LDS: written by phase B, consumed
// as WMMA A-fragments (ds_load) by phase A of the next step.
// ---------------------------------------------------------------------------
__global__ __launch_bounds__(1024) void lstm_recurrent(
    const __bf16* __restrict__ embT,  // [8192, 256]
    const __bf16* __restrict__ WxT,   // [2048, 256]
    const __bf16* __restrict__ WhT,   // [2048, 512]
    const float* __restrict__ b_i, const float* __restrict__ b_f,
    const float* __restrict__ b_c, const float* __restrict__ b_o,
    __bf16* __restrict__ hsbf,        // [8192, 512] all hidden states (bf16)
    float* __restrict__ gbuf,         // [64, 2048] gate pre-activations
    float* __restrict__ h_out,        // [64, 512] fp32 (final h, in d_out)
    float* __restrict__ c_out)        // [64, 512] fp32 (cell state, in d_out)
{
    __shared__ __align__(16) __bf16 hlds[BB * HH];   // 64 KB of the 320 KB WGP LDS

    const int tid  = threadIdx.x;
    const int wave = tid >> 5;  // wave32

    for (int t = 0; t < TT; ++t) {
        // prefetch next timestep's embedding panel (64x256 bf16 = 32 KB)
        if (t + 1 < TT)
            __builtin_prefetch(embT + (size_t)(t + 1) * BB * EE + tid * 32, 0, 1);

        // ---- phase A: fused 4-gate GEMM, 4x128 = 512 16x16 tiles, 16/wave
        const __bf16* At = embT + (size_t)t * BB * EE;
        for (int job = wave; job < (BB / 16) * (G4 / 16); job += 32) {
            const int mt = job >> 7;
            const int nt = job & 127;
            const int m0 = mt * 16, n0 = nt * 16;
            v8f acc = {};
            for (int k0 = 0; k0 < EE; k0 += 32) {              // x-projection
                v16bf a = load_a_frag(At, EE, m0, k0);
                v16bf b = load_b_frag(WxT, EE, k0, n0);
                acc = wmma_bf16(a, b, acc);
            }
            if (t > 0) {                                       // h-projection (LDS A)
                for (int k0 = 0; k0 < HH; k0 += 32) {
                    v16bf a = load_a_frag(hlds, HH, m0, k0);   // ds_load fragments
                    v16bf b = load_b_frag(WhT, HH, k0, n0);
                    acc = wmma_bf16(a, b, acc);
                }
            }
            store_c_tile(gbuf, G4, m0, n0, acc);
        }
        __threadfence();
        __syncthreads();

        // ---- phase B: LSTM cell elementwise (64*512 elems, 32/thread)
        for (int i = tid; i < BB * HH; i += 1024) {
            const int  bb = i >> 9;
            const int  n  = i & 511;
            const float* g = gbuf + (size_t)bb * G4;
            const float gi = g[n]        + b_i[n];
            const float gf = g[n +  512] + b_f[n];
            const float gc = g[n + 1024] + b_c[n];
            const float go = g[n + 1536] + b_o[n];
            const float it = 1.f / (1.f + __expf(-gi));
            const float ft = 1.f / (1.f + __expf(-gf));
            const float gt = tanhf(gc);
            const float ot = 1.f / (1.f + __expf(-go));
            const float cp = (t == 0) ? 0.f : c_out[i];
            const float c  = ft * cp + it * gt;
            const float h  = ot * tanhf(c);
            c_out[i] = c;
            h_out[i] = h;
            const __bf16 hb = (__bf16)h;
            hlds[i] = hb;
            hsbf[(size_t)t * BB * HH + i] = hb;
        }
        __threadfence();
        __syncthreads();
    }
}

// ---------------------------------------------------------------------------
// Output projection: out[8192,10000] = hs[8192,512] @ W[512,10000] + b.
// Block = 8 waves; block tile = 128 rows x 80 cols.  All 8 waves share the
// same 80x32 bf16 B panel per K-step -> double-buffered LDS staging via
// async global->LDS copies overlapped with WMMA (ASYNCcnt pipelining).
// Grid: 64 M-blocks x 125 strips = 8000 blocks.
// ---------------------------------------------------------------------------
__global__ __launch_bounds__(256) void output_gemm(
    const __bf16* __restrict__ hsbf,  // [8192, 512]
    const __bf16* __restrict__ WoT,   // [10000, 512] transposed bf16
    const float* __restrict__ bias,   // [10000]
    float* __restrict__ out)          // [8192, 10000]
{
    __shared__ __align__(16) __bf16 bstage[2][80 * 32];  // 2 x 5 KB panels

    const int tid    = threadIdx.x;
    const int wave   = tid >> 5;
    const int mblock = blockIdx.x / 125;       // 0..63
    const int strip  = blockIdx.x - mblock * 125;
    const int m0 = (mblock * 8 + wave) * 16;   // this wave's 16 rows
    const int n0 = strip * 80;                 // block's 80 cols

    // stage WoT rows [n0,n0+80), K [k0,k0+32) into bstage[buf]
    // 80 rows x 64 B = 320 16-byte transfers; waves 0-4 (160 threads) x2 each
    auto stage = [&](int k0, int buf) {
        if (tid < 160) {
#pragma unroll
            for (int s = 0; s < 2; ++s) {
                const int g   = tid * 2 + s;   // 0..319
                const int row = g >> 2;        // 0..79
                const int ch  = g & 3;         // 16B chunk in row
                copy16_to_lds(&bstage[buf][row * 32 + ch * 8],
                              WoT + (size_t)(n0 + row) * HH + k0 + ch * 8);
            }
        }
    };

    stage(0, 0);
    v8f acc[5] = {{}, {}, {}, {}, {}};

    for (int ks = 0; ks < HH / 32; ++ks) {     // 16 K-steps
        const int buf = ks & 1;
        if (ks + 1 < HH / 32) {
            stage((ks + 1) * 32, buf ^ 1);     // prefetch next panel
            async_wait_le(2);                  // current panel has landed
        } else {
            async_wait_le(0);
        }
        __syncthreads();                       // panel visible to all waves

        v16bf a = load_a_frag(hsbf, HH, m0, ks * 32);

        // 5 B fragments from the shared LDS panel (ds_load_b128 x2 each)
        const int lane = tid & 31;
        const int koff = (lane >> 4) << 4;
        v16bf bf[5];
#pragma unroll
        for (int j = 0; j < 5; ++j) {
            const __bf16* p = &bstage[buf][(j * 16 + (lane & 15)) * 32 + koff];
            Frag16 f;
            f.q[0] = *reinterpret_cast<const uint4*>(p);
            f.q[1] = *reinterpret_cast<const uint4*>(p + 8);
            bf[j] = f.v;
        }

        // same A operand 5x back-to-back: set reuse-A hint on the first four
        acc[0] = __builtin_amdgcn_wmma_f32_16x16x32_bf16(false, a, false, bf[0], (short)0, acc[0], true,  false);
        acc[1] = __builtin_amdgcn_wmma_f32_16x16x32_bf16(false, a, false, bf[1], (short)0, acc[1], true,  false);
        acc[2] = __builtin_amdgcn_wmma_f32_16x16x32_bf16(false, a, false, bf[2], (short)0, acc[2], true,  false);
        acc[3] = __builtin_amdgcn_wmma_f32_16x16x32_bf16(false, a, false, bf[3], (short)0, acc[3], true,  false);
        acc[4] = __builtin_amdgcn_wmma_f32_16x16x32_bf16(false, a, false, bf[4], (short)0, acc[4], false, false);

        __syncthreads();                       // done reading buf before overwrite
    }

    const int lane = tid & 31;
#pragma unroll
    for (int j = 0; j < 5; ++j) {
        const int col = n0 + j * 16 + (lane & 15);
        const int rb  = m0 + ((lane >> 4) << 3);
        const float bv = bias[col];
#pragma unroll
        for (int r = 0; r < 8; ++r)
            out[(size_t)(rb + r) * VV + col] = acc[j][r] + bv;
    }
}

// ---------------------------------------------------------------------------
extern "C" void kernel_launch(void* const* d_in, const int* in_sizes, int n_in,
                              void* d_out, int out_size, void* d_ws, size_t ws_size,
                              hipStream_t stream) {
    (void)in_sizes; (void)n_in; (void)out_size; (void)ws_size;

    const int*   ids  = (const int*)d_in[0];
    const float* emb  = (const float*)d_in[1];
    const float* W_xi = (const float*)d_in[2];
    const float* W_hi = (const float*)d_in[3];
    const float* b_i  = (const float*)d_in[4];
    const float* W_xf = (const float*)d_in[5];
    const float* W_hf = (const float*)d_in[6];
    const float* b_f  = (const float*)d_in[7];
    const float* W_xc = (const float*)d_in[8];
    const float* W_hc = (const float*)d_in[9];
    const float* b_c  = (const float*)d_in[10];
    const float* W_xo = (const float*)d_in[11];
    const float* W_ho = (const float*)d_in[12];
    const float* b_o  = (const float*)d_in[13];
    const float* W    = (const float*)d_in[14];
    const float* b    = (const float*)d_in[15];

    // workspace carve-up (~26 MB)
    char* ws = (char*)d_ws;
    __bf16* embT = (__bf16*)ws; ws += (size_t)TT * BB * EE * 2;  // 4 MB
    __bf16* WxT  = (__bf16*)ws; ws += (size_t)G4 * EE * 2;       // 1 MB
    __bf16* WhT  = (__bf16*)ws; ws += (size_t)G4 * HH * 2;       // 2 MB
    __bf16* WoT  = (__bf16*)ws; ws += (size_t)VV * HH * 2;       // 10 MB
    __bf16* hsbf = (__bf16*)ws; ws += (size_t)TT * BB * HH * 2;  // 8 MB
    float*  gbuf = (float*)ws;  ws += (size_t)BB * G4 * 4;       // 512 KB

    float* out   = (float*)d_out;
    float* h_out = out + (size_t)TT * BB * VV;   // [64,512]
    float* c_out = h_out + BB * HH;              // [64,512]

    // ---- prep: gather + bf16 transposes
    {
        const int n = TT * BB * EE;
        gather_embed_bf16<<<(n + 255) / 256, 256, 0, stream>>>(ids, emb, embT);
    }
    {
        const int nx = EE * HH, nh = HH * HH;
        transpose_bf16<<<(nx + 255) / 256, 256, 0, stream>>>(W_xi, WxT + 0 * (size_t)HH * EE, EE, HH);
        transpose_bf16<<<(nx + 255) / 256, 256, 0, stream>>>(W_xf, WxT + 1 * (size_t)HH * EE, EE, HH);
        transpose_bf16<<<(nx + 255) / 256, 256, 0, stream>>>(W_xc, WxT + 2 * (size_t)HH * EE, EE, HH);
        transpose_bf16<<<(nx + 255) / 256, 256, 0, stream>>>(W_xo, WxT + 3 * (size_t)HH * EE, EE, HH);
        transpose_bf16<<<(nh + 255) / 256, 256, 0, stream>>>(W_hi, WhT + 0 * (size_t)HH * HH, HH, HH);
        transpose_bf16<<<(nh + 255) / 256, 256, 0, stream>>>(W_hf, WhT + 1 * (size_t)HH * HH, HH, HH);
        transpose_bf16<<<(nh + 255) / 256, 256, 0, stream>>>(W_hc, WhT + 2 * (size_t)HH * HH, HH, HH);
        transpose_bf16<<<(nh + 255) / 256, 256, 0, stream>>>(W_ho, WhT + 3 * (size_t)HH * HH, HH, HH);
        const int nw = HH * VV;
        transpose_bf16<<<(nw + 255) / 256, 256, 0, stream>>>(W, WoT, HH, VV);
    }

    // ---- recurrence: one persistent workgroup (32 wave32s on one WGP)
    lstm_recurrent<<<1, 1024, 0, stream>>>(embT, WxT, WhT, b_i, b_f, b_c, b_o,
                                           hsbf, gbuf, h_out, c_out);

    // ---- output projection: 8000 blocks x 8 waves, LDS-staged B panels
    output_gemm<<<8000, 256, 0, stream>>>(hsbf, WoT, b, out);
}
